// Mamba_v1_27865747817195
// MI455X (gfx1250) — matrix operand order repaired and checked by reading, not compiled, not run on hardware
//
#include <hip/hip_runtime.h>
#include <hip/hip_bf16.h>

#define PP   8192
#define LSEQ 6
#define HH   32
#define DI   64
#define NS   64
#define RNK  2
#define NROWS (PP*LSEQ)

typedef __attribute__((ext_vector_type(16))) _Float16 v16h;
typedef __attribute__((ext_vector_type(8)))  float    v8f;
typedef unsigned int u32x4 __attribute__((ext_vector_type(4)));
typedef unsigned int u32x8 __attribute__((ext_vector_type(8)));

__device__ __forceinline__ v8f wmma16(v16h a, v16h b, v8f c) {
  return __builtin_amdgcn_wmma_f32_16x16x32_f16(false, a, false, b, (short)0, c, false, false);
}
__device__ __forceinline__ float sigm_(float x){ return 1.f/(1.f+__expf(-x)); }
__device__ __forceinline__ float silu_(float x){ return x*sigm_(x); }
__device__ __forceinline__ float softplus_(float x){ return (x>20.f)? x : __logf(1.f+__expf(x)); }

// ---------------- Kernel 1: feature gather + fc1 + relu ----------------
__device__ __forceinline__ float feat_(const float* xs,const float* xd,const float* xp,
                                       const float* xt,int p,int l,int j){
  if (j<22) return xs[p*22+j];
  if (j<34) return xd[(l*PP+p)*12 + (j-22)];
  if (j<41) return xp[(l*PP+p)*7  + (j-34)];
  if (j<45) return xt[l*4 + (j-41)];
  return 0.f;
}

__global__ void __launch_bounds__(256) k_fc1(const float* __restrict__ xs, const float* __restrict__ xd,
     const float* __restrict__ xp, const float* __restrict__ xt,
     const float* __restrict__ w, const float* __restrict__ bias, float* __restrict__ X1) {
  const int wv = threadIdx.x>>5, lane = threadIdx.x&31;
  const int m0 = lane&15, hi = lane>>4;
  const int rowBase = (blockIdx.x*8 + wv)*16;
  const int r = rowBase + m0;
  const int p = r/LSEQ, l = r - p*LSEQ;
  v16h a0,a1;
  #pragma unroll
  for (int e=0;e<16;++e){
    int k = e + (hi<<4);                        // 16-bit A 16x32 layout (contiguous k, matches B)
    a0[e] = (_Float16)feat_(xs,xd,xp,xt,p,l,k);
    a1[e] = (_Float16)feat_(xs,xd,xp,xt,p,l,k+32);
  }
  #pragma unroll
  for (int nt=0; nt<2; ++nt){
    const int n = nt*16 + m0;
    v16h b0,b1;
    #pragma unroll
    for (int e=0;e<16;++e){
      int k = e + (hi<<4);                      // 16-bit B 32x16 layout
      b0[e] = (_Float16)w[n*45+k];
      int k2 = k+32;
      b1[e] = (k2<45)? (_Float16)w[n*45+k2] : (_Float16)0.f;
    }
    v8f c = {};
    c = wmma16(a0,b0,c);
    c = wmma16(a1,b1,c);
    #pragma unroll
    for (int i=0;i<8;++i){
      int mr = i + (hi<<3);
      float v = c[i] + bias[n];
      v = v>0.f ? v : 0.f;
      X1[(long)(rowBase+mr)*HH + n] = v;
    }
  }
}

// ---------------- Kernel 2: LN + Mamba block + residual (one wave per sequence) ----------------
__global__ void __launch_bounds__(128) k_mamba(const float* __restrict__ Xin, float* __restrict__ Xout,
    const float* __restrict__ lnw, const float* __restrict__ lnb,
    const float* __restrict__ in_w, const float* __restrict__ conv_w, const float* __restrict__ conv_b,
    const float* __restrict__ xproj_w, const float* __restrict__ dt_w, const float* __restrict__ dt_b,
    const float* __restrict__ A_log, const float* __restrict__ Dp, const float* __restrict__ out_w) {
  // A_s final layout: float2 pairs, pair q = n*32+lane -> {Ascaled[d=lane][n], Ascaled[d=lane+32][n]}
  __shared__ __align__(16) float A_s[NS*DI];
  __shared__ __align__(16) float xb  [4][LSEQ*HH];     // post-LN x
  __shared__ __align__(16) float xzb [4][LSEQ*2*DI];   // in_proj out (xc_pre | z)
  __shared__ __align__(16) float xcb [4][LSEQ*DI];     // conv+silu
  __shared__ __align__(16) float dblb[4][LSEQ*132];    // dt(2) | interleaved {B,C} pairs (128)
  __shared__ __align__(16) float yb  [4][LSEQ*DI];     // scan output (gated)

  const int tid = threadIdx.x;
  const int wv = tid>>5, lane = tid&31;
  const int m0 = lane&15, hi = lane>>4;
  const int s = blockIdx.x*4 + wv;

  __builtin_prefetch(in_w, 0, 3);
  __builtin_prefetch(xproj_w, 0, 3);

  // ---- stage 0: TDM async copy of A_log (16KB) into LDS, then pack in place ----
  if (wv==0){
    unsigned ldsAddr = (unsigned)(unsigned long long)(uintptr_t)A_s; // low 32 bits = LDS offset
    unsigned long long ga = (unsigned long long)(uintptr_t)A_log;
    u32x4 g0;
    g0[0] = 1u;                                               // count=1 (valid user descriptor)
    g0[1] = ldsAddr;                                          // lds_addr
    g0[2] = (unsigned)ga;                                     // global_addr[31:0]
    g0[3] = (unsigned)((ga>>32) & 0x01FFFFFFu) | 0x80000000u; // global_addr[56:32] | type=2
    u32x8 g1;
    g1[0] = (2u<<16);                  // workgroup_mask=0, data_size=4B
    g1[1] = (4096u & 0xFFFFu) << 16;   // tensor_dim0[15:0] -> bits 63:48
    g1[2] = (1u<<16);                  // tensor_dim0[31:16]=0, tensor_dim1=1
    g1[3] = (4096u & 0xFFFFu) << 16;   // tensor_dim1 hi=0, tile_dim0=4096
    g1[4] = 1u;                        // tile_dim1=1, tile_dim2=0
    g1[5] = 4096u;                     // tensor_dim0_stride[31:0]
    g1[6] = 0u;                        // stride hi, tensor_dim1_stride lo
    g1[7] = 0u;
    asm volatile("tensor_load_to_lds %0, %1" :: "s"(g0), "s"(g1) : "memory");
    __builtin_amdgcn_s_wait_tensorcnt(0);
  }
  __syncthreads();
  {
    // in-place: A_s currently raw A_log[d*64+n]; repack to float2 pairs with -log2(e)*exp()
    float tv[32];
    #pragma unroll
    for (int j=0;j<32;++j) tv[j] = A_s[tid*32 + j];
    __syncthreads();
    #pragma unroll
    for (int j=0;j<32;++j){
      int i = tid*32 + j;
      int d = i>>6, n = i&63;
      A_s[(n*32 + (d&31))*2 + (d>>5)] = -1.4426950408889634f * __expf(tv[j]);
    }
  }

  // ---- stage 1: LayerNorm (lane == h) ----
  #pragma unroll
  for (int t=0;t<LSEQ;++t){
    float v = Xin[((long)s*LSEQ+t)*HH + lane];
    float su = v;
    #pragma unroll
    for (int off=16; off; off>>=1) su += __shfl_xor(su, off, 32);
    float mu = su*(1.f/32.f);
    float dv = v-mu;
    float vq = dv*dv;
    #pragma unroll
    for (int off=16; off; off>>=1) vq += __shfl_xor(vq, off, 32);
    float var = vq*(1.f/32.f);
    xb[wv][t*HH+lane] = dv*rsqrtf(var+1e-5f)*lnw[lane] + lnb[lane];
  }
  __syncthreads();

  // ---- stage 2: in_proj (6x32)@(32x128) via WMMA ----
  {
    v16h a;
    #pragma unroll
    for (int e=0;e<16;++e){
      int k = e + (hi<<4);
      a[e] = (m0<LSEQ)? (_Float16)xb[wv][m0*HH+k] : (_Float16)0.f;
    }
    #pragma unroll
    for (int nt=0; nt<8; ++nt){
      int n = nt*16 + m0;
      v16h bfr;
      #pragma unroll
      for (int e=0;e<16;++e){
        int k = e + (hi<<4);
        bfr[e] = (_Float16)in_w[n*HH+k];
      }
      v8f c = {};
      c = wmma16(a,bfr,c);
      #pragma unroll
      for (int i=0;i<8;++i){
        int mr = i + (hi<<3);
        if (mr<LSEQ) xzb[wv][mr*128 + n] = c[i];
      }
    }
  }
  __syncthreads();

  // ---- stage 3: causal depthwise conv (K=4) + silu (lane -> d, d+32) ----
  {
    const int d0 = lane, d1 = lane+32;
    #pragma unroll
    for (int t=0;t<LSEQ;++t){
      float acc0 = conv_b[d0], acc1 = conv_b[d1];
      #pragma unroll
      for (int k=0;k<4;++k){
        int tt = t + k - 3;
        if (tt>=0){
          acc0 += xzb[wv][tt*128 + d0] * conv_w[d0*4+k];
          acc1 += xzb[wv][tt*128 + d1] * conv_w[d1*4+k];
        }
      }
      xcb[wv][t*DI+d0] = silu_(acc0);
      xcb[wv][t*DI+d1] = silu_(acc1);
    }
  }
  __syncthreads();

  // ---- stage 4: xproj (6x64)@(64x130) via WMMA; scatter B/C interleaved ----
  {
    v16h a0,a1;
    #pragma unroll
    for (int e=0;e<16;++e){
      int k = e + (hi<<4);
      a0[e] = (m0<LSEQ)? (_Float16)xcb[wv][m0*DI+k]    : (_Float16)0.f;
      a1[e] = (m0<LSEQ)? (_Float16)xcb[wv][m0*DI+k+32] : (_Float16)0.f;
    }
    #pragma unroll
    for (int nt=0; nt<9; ++nt){
      int n = nt*16 + m0;
      bool nv = (n < RNK + 2*NS);
      // column permutation: dt cols 0..1 keep; B col (2..65) -> 2+2*(n-2); C col (66..129) -> 3+2*(n-66)
      int cp = (n<2)? n : ((n<66)? (2 + 2*(n-2)) : (3 + 2*(n-66)));
      v16h b0,b1;
      #pragma unroll
      for (int e=0;e<16;++e){
        int k = e + (hi<<4);
        b0[e] = nv? (_Float16)xproj_w[n*DI+k]    : (_Float16)0.f;
        b1[e] = nv? (_Float16)xproj_w[n*DI+k+32] : (_Float16)0.f;
      }
      v8f c = {};
      c = wmma16(a0,b0,c);
      c = wmma16(a1,b1,c);
      #pragma unroll
      for (int i=0;i<8;++i){
        int mr = i + (hi<<3);
        if (mr<LSEQ && nv) dblb[wv][mr*132 + cp] = c[i];
      }
    }
  }
  __syncthreads();

  // ---- stage 5: softplus(delta) + selective scan; h in 128 VGPRs/lane,
  //      2x ds_load_b64 per (t,n): {A_d0,A_d1} pair + {B,C} pair ----
  {
    const int d0 = lane, d1 = lane+32;
    const float dw00=dt_w[d0*2], dw01=dt_w[d0*2+1];
    const float dw10=dt_w[d1*2], dw11=dt_w[d1*2+1];
    const float db0=dt_b[d0], db1=dt_b[d1];
    const float Dp0=Dp[d0],  Dp1=Dp[d1];
    float h0[NS], h1[NS];
    #pragma unroll
    for (int n=0;n<NS;++n){ h0[n]=0.f; h1[n]=0.f; }
    const float2* Apk = (const float2*)A_s;
    #pragma unroll
    for (int t=0;t<LSEQ;++t){
      float t0 = dblb[wv][t*132+0], t1 = dblb[wv][t*132+1];
      float de0 = softplus_(t0*dw00 + t1*dw01 + db0);
      float de1 = softplus_(t0*dw10 + t1*dw11 + db1);
      float xc0 = xcb[wv][t*DI+d0], xc1 = xcb[wv][t*DI+d1];
      float dx0 = de0*xc0, dx1 = de1*xc1;
      float acc0=0.f, acc1=0.f;
      const float2* bcp = (const float2*)&dblb[wv][t*132+2];
      #pragma unroll
      for (int n=0;n<NS;++n){
        float2 Ap = Apk[n*32 + lane];   // {A'[d0][n], A'[d1][n]}, A' = -log2e*exp(A_log)
        float2 bc = bcp[n];             // {B[n], C[n]} broadcast
        h0[n] = h0[n]*__builtin_amdgcn_exp2f(de0*Ap.x) + dx0*bc.x;
        h1[n] = h1[n]*__builtin_amdgcn_exp2f(de1*Ap.y) + dx1*bc.x;
        acc0 += h0[n]*bc.y;
        acc1 += h1[n]*bc.y;
      }
      float z0 = xzb[wv][t*128+DI+d0], z1 = xzb[wv][t*128+DI+d1];
      yb[wv][t*DI+d0] = (acc0 + xc0*Dp0) * silu_(z0);
      yb[wv][t*DI+d1] = (acc1 + xc1*Dp1) * silu_(z1);
    }
  }
  __syncthreads();

  // ---- stage 6: out_proj (6x64)@(64x32) via WMMA + residual ----
  {
    v16h a0,a1;
    #pragma unroll
    for (int e=0;e<16;++e){
      int k = e + (hi<<4);
      a0[e] = (m0<LSEQ)? (_Float16)yb[wv][m0*DI+k]    : (_Float16)0.f;
      a1[e] = (m0<LSEQ)? (_Float16)yb[wv][m0*DI+k+32] : (_Float16)0.f;
    }
    #pragma unroll
    for (int nt=0; nt<2; ++nt){
      int n = nt*16 + m0;
      v16h b0,b1;
      #pragma unroll
      for (int e=0;e<16;++e){
        int k = e + (hi<<4);
        b0[e] = (_Float16)out_w[n*DI+k];
        b1[e] = (_Float16)out_w[n*DI+k+32];
      }
      v8f c = {};
      c = wmma16(a0,b0,c);
      c = wmma16(a1,b1,c);
      #pragma unroll
      for (int i=0;i<8;++i){
        int mr = i + (hi<<3);
        if (mr<LSEQ){
          long r = (long)s*LSEQ + mr;
          Xout[r*HH + n] = c[i] + Xin[r*HH + n];
        }
      }
    }
  }
}

// ---------------- Kernel 3: last-step heads ----------------
__global__ void __launch_bounds__(256) k_heads(const float* __restrict__ X,
    const float* __restrict__ fc2_w, const float* __restrict__ fc2_b,
    const float* __restrict__ fc3_w, const float* __restrict__ fc3_b,
    const float* __restrict__ fc4_w, const float* __restrict__ fc4_b,
    float* __restrict__ out) {
  __shared__ float ybuf[8][16][HH];
  const int wv = threadIdx.x>>5, lane = threadIdx.x&31;
  const int m0 = lane&15, hi = lane>>4;
  const int prBase = blockIdx.x*128 + wv*16;
  {
    v16h a;
    #pragma unroll
    for (int e=0;e<16;++e){
      int k = e + (hi<<4);
      a[e] = (_Float16)X[((long)(prBase+m0)*LSEQ + (LSEQ-1))*HH + k];
    }
    #pragma unroll
    for (int nt=0; nt<2; ++nt){
      int n = nt*16+m0;
      v16h bfr;
      #pragma unroll
      for (int e=0;e<16;++e){
        int k = e + (hi<<4);
        bfr[e] = (_Float16)fc2_w[n*HH + k];
      }
      v8f c = {};
      c = wmma16(a,bfr,c);
      #pragma unroll
      for (int i=0;i<8;++i){
        int mr = i + (hi<<3);
        float v = c[i] + fc2_b[n];
        v = (v>0.f)? v : 0.2f*v;          // leaky_relu(0.2)
        ybuf[wv][mr][n] = v;
      }
    }
  }
  __syncthreads();
  {
    const int pr = prBase + m0;
    float xv[HH];
    #pragma unroll
    for (int h=0; h<HH; ++h) xv[h] = ybuf[wv][m0][h];
    if (hi==0){
      #pragma unroll
      for (int o=0;o<7;++o){
        float acc = fc3_b[o];
        #pragma unroll
        for (int h=0;h<HH;++h) acc += xv[h]*fc3_w[o*HH+h];
        out[(long)pr*7 + o] = acc;
      }
    } else {
      #pragma unroll
      for (int o=0;o<3;++o){
        float acc = fc4_b[o];
        #pragma unroll
        for (int h=0;h<HH;++h) acc += xv[h]*fc4_w[o*HH+h];
        out[(long)PP*7 + (long)pr*3 + o] = acc;
      }
    }
  }
  if (blockIdx.x==0 && threadIdx.x==0){
    out[(long)PP*7 + (long)PP*3]     = 0.f;   // scalar zero #1
    out[(long)PP*7 + (long)PP*3 + 1] = 0.f;   // scalar zero #2
  }
}

extern "C" void kernel_launch(void* const* d_in, const int* in_sizes, int n_in,
                              void* d_out, int out_size, void* d_ws, size_t ws_size,
                              hipStream_t stream) {
  (void)in_sizes; (void)n_in; (void)out_size; (void)ws_size;
  const float* xs    = (const float*)d_in[0];
  const float* xd    = (const float*)d_in[1];
  const float* xp    = (const float*)d_in[2];
  const float* xt    = (const float*)d_in[3];
  const float* fc1_w = (const float*)d_in[4];
  const float* fc1_b = (const float*)d_in[5];
  const float* ln1_w = (const float*)d_in[6];
  const float* ln1_b = (const float*)d_in[7];
  const float* ln2_w = (const float*)d_in[8];
  const float* ln2_b = (const float*)d_in[9];
  const float* fc2_w = (const float*)d_in[10];
  const float* fc2_b = (const float*)d_in[11];
  const float* fc3_w = (const float*)d_in[12];
  const float* fc3_b = (const float*)d_in[13];
  const float* fc4_w = (const float*)d_in[14];
  const float* fc4_b = (const float*)d_in[15];
  const float* m1[9]; const float* m2[9];
  for (int i=0;i<9;++i){ m1[i] = (const float*)d_in[16+i]; m2[i] = (const float*)d_in[25+i]; }
  // m[k]: 0=in_w 1=conv_w 2=conv_b 3=xproj_w 4=dt_w 5=dt_b 6=A_log 7=D 8=out_w

  float* X1 = (float*)d_ws;                       // 49152*32 f32 = 6 MB
  float* X2 = X1 + (size_t)NROWS*HH;              // another 6 MB
  float* X3 = X1;                                 // reuse X1 for block-2 output

  k_fc1  <<<384, 256, 0, stream>>>(xs, xd, xp, xt, fc1_w, fc1_b, X1);
  k_mamba<<<2048,128, 0, stream>>>(X1, X2, ln1_w, ln1_b,
      m1[0], m1[1], m1[2], m1[3], m1[4], m1[5], m1[6], m1[7], m1[8]);
  k_mamba<<<2048,128, 0, stream>>>(X2, X3, ln2_w, ln2_b,
      m2[0], m2[1], m2[2], m2[3], m2[4], m2[5], m2[6], m2[7], m2[8]);
  k_heads<<<64,  256, 0, stream>>>(X3, fc2_w, fc2_b, fc3_w, fc3_b, fc4_w, fc4_b,
      (float*)d_out);
}
